// MGCDModule_7404523619105
// MI455X (gfx1250) — matrix operand order repaired and checked by reading, not compiled, not run on hardware
//
#include <hip/hip_runtime.h>
#include <hip/hip_bf16.h>

typedef __attribute__((ext_vector_type(16))) _Float16 v16h;
typedef __attribute__((ext_vector_type(8)))  float    v8f;

#define WAVES_PER_BLOCK 8
#define NGROUPS 4

struct ConvDesc {
  int B, C, CR, HA, WA, HB, WB;   // C = GEMM channels, CR = real input channels (concat wrap)
  int OC, S, P;
  int HB2, WB2;                   // output support dims (1 for plain 2D conv)
  int Mtotal, Ktot, Kp;           // Kp = Ktot padded to multiple of 32 (weight row pitch)
  int relu, accumulate;
};

__device__ __forceinline__ void decode_m_q(int m, const ConvDesc& d,
                                           int& b, int& hb2, int& wb2, int& oy, int& ox) {
  ox = m % d.WA; int t = m / d.WA;
  oy = t % d.HA; t /= d.HA;
  wb2 = t % d.WB2; t /= d.WB2;
  hb2 = t % d.HB2; b = t / d.HB2;
}
__device__ __forceinline__ void decode_m_s(int m, const ConvDesc& d,
                                           int& b, int& ha, int& wa, int& oyb, int& oxb) {
  oxb = m % d.WB2; int t = m / d.WB2;
  oyb = t % d.HB2; t /= d.HB2;
  wa = t % d.WA; t /= d.WA;
  ha = t % d.HA; b = t / d.HA;
}

// Implicit-GEMM conv on v_wmma_f32_16x16x32_f16.
// MODE 0: conv over query dims (ha,wa), stride 1, input pruned on support grid by S.
//         (With HB=WB=HB2=WB2=1 this is a plain NCHW 3x3 conv for the decoder.)
// MODE 1: conv over support dims (hb,wb) with stride S; accumulates into y (out1 + out2).
// KSZ is the spatial kernel size (3 or 5) -> all im2col decode divides are by constants.
// A fragment (16x32 f16): lane&15 = M row, k = (lane>>4)*16 + i, i.e. K halves {0..15}/{16..31} (same order as B).
// B fragment (32x16 f16): lane&15 = N col, k = (lane>>4)*16 + i  => 16 contiguous f16 =
//   one aligned 32B vector load from the zero-padded packed weight matrix.
// C/D (16x16 f32): m = vgpr + 8*(lane>>4), n = lane&15.
template<int NT, int MODE, int KSZ>
__global__ __launch_bounds__(256) void conv_wmma(const float* __restrict__ x,
                                                 const _Float16* __restrict__ w,
                                                 const float* __restrict__ bias,
                                                 float* __restrict__ y, ConvDesc d)
{
  const int lane = threadIdx.x & 31;
  const int wave = threadIdx.x >> 5;
  const int mtile = (blockIdx.x * WAVES_PER_BLOCK + wave) * 16;
  if (mtile >= d.Mtotal) return;   // wave-uniform: EXEC stays all-ones for WMMA

  constexpr int KK = KSZ * KSZ;

  int am = mtile + (lane & 15);
  const bool mval = am < d.Mtotal;
  if (!mval) am = d.Mtotal - 1;
  int b, u0, u1, v0, v1;
  if (MODE == 0) decode_m_q(am, d, b, u0, u1, v0, v1);  // u=(hb2,wb2), v=(oy,ox)
  else           decode_m_s(am, d, b, u0, u1, v0, v1);  // u=(ha,wa),  v=(oyb,oxb)

  const int kbaseA = (lane >> 4) * 16;
  const int nB     = lane & 15;
  const int kbaseB = (lane >> 4) * 16;

  v8f acc[NT] = {};

  const int Ksteps = d.Kp >> 5;
  for (int ks = 0; ks < Ksteps; ++ks) {
    const int k0 = ks << 5;
    if (k0 + 32 < d.Kp)  // hint the weight stream (global_prefetch_b8)
      __builtin_prefetch(w + (size_t)nB * d.Kp + k0 + 32, 0, 1);

    // ---- gather A fragment (im2col on the fly, f32 -> f16 in-register) ----
    v16h a;
#pragma unroll
    for (int i = 0; i < 16; ++i) {
      const int k = k0 + kbaseA + i;
      float v = 0.0f;
      if (k < d.Ktot && mval) {
        int c = k / KK; int r = k - c * KK;        // constant-divisor: mul+shift
        int ky = r / KSZ; int kx = r - ky * KSZ;
        int ce = (c >= d.CR) ? (c - d.CR) : c;     // channel-concat wrap
        if (MODE == 0) {
          int iy = v0 + ky - d.P, ix = v1 + kx - d.P;
          if ((unsigned)iy < (unsigned)d.HA && (unsigned)ix < (unsigned)d.WA) {
            int hb = u0 * d.S, wb = u1 * d.S;      // pruned support sample
            long addr = ((((long)(b * d.CR + ce) * d.HA + iy) * d.WA + ix) * d.HB + hb) * d.WB + wb;
            v = x[addr];
          }
        } else {
          int iy = v0 * d.S + ky - d.P, ix = v1 * d.S + kx - d.P;
          if ((unsigned)iy < (unsigned)d.HB && (unsigned)ix < (unsigned)d.WB) {
            long addr = ((((long)(b * d.CR + ce) * d.HA + u0) * d.WA + u1) * d.HB + iy) * d.WB + ix;
            v = x[addr];
          }
        }
      }
      a[i] = (_Float16)v;
    }

    // ---- B fragments: one aligned 32-byte vector load per tile (padded weights) ----
#pragma unroll
    for (int t = 0; t < NT; ++t) {
      const int oc = t * 16 + nB;                  // rows >= OC are zero-padded
      v16h bf = *(const v16h*)(w + (size_t)oc * d.Kp + k0 + kbaseB);
      acc[t] = __builtin_amdgcn_wmma_f32_16x16x32_f16(
          false, a, false, bf, (short)0, acc[t], false, false);
    }
  }

  // ---- scatter D (bias, optional accumulate with conv1 result, optional relu) ----
  const int mrow0 = (lane >> 4) * 8;
#pragma unroll
  for (int t = 0; t < NT; ++t) {
    const int oc = t * 16 + nB;
    if (oc >= d.OC) continue;
    const float bv = bias[oc];
#pragma unroll
    for (int j = 0; j < 8; ++j) {
      const int mm = mtile + mrow0 + j;
      if (mm >= d.Mtotal) continue;
      int ob, o0, o1, o2, o3;
      long oaddr;
      if (MODE == 0) {
        decode_m_q(mm, d, ob, o0, o1, o2, o3);
        oaddr = ((((long)(ob * d.OC + oc) * d.HA + o2) * d.WA + o3) * d.HB2 + o0) * d.WB2 + o1;
      } else {
        decode_m_s(mm, d, ob, o0, o1, o2, o3);
        oaddr = ((((long)(ob * d.OC + oc) * d.HA + o0) * d.WA + o1) * d.HB2 + o2) * d.WB2 + o3;
      }
      float v = acc[t][j] + bv;
      if (d.accumulate) v += y[oaddr];
      if (d.relu) v = fmaxf(v, 0.0f);
      y[oaddr] = v;
    }
  }
}

// Pack [OC, Ktot] f32 weights into zero-padded [OCp, Kp] f16 (row pitch Kp, mult of 32).
__global__ void pack_w_f16(const float* __restrict__ src, _Float16* __restrict__ dst,
                           int OC, int Ktot, int Kp, int total) {
  int i = blockIdx.x * blockDim.x + threadIdx.x;
  if (i >= total) return;
  int oc = i / Kp, k = i - oc * Kp;
  float v = (oc < OC && k < Ktot) ? src[(size_t)oc * Ktot + k] : 0.0f;
  dst[i] = (_Float16)v;
}

// GroupNorm stats: one block per (batch, group); group channels are contiguous, so the
// reduction range is contiguous. Deterministic LDS tree reduction (no float atomics).
__global__ __launch_bounds__(256) void gn_stats(const float* __restrict__ x, float* __restrict__ stats,
                                                int C, int SP, int chPerG, int G) {
  const int bg = blockIdx.x;
  const int b = bg / G, g = bg % G;
  const size_t base = ((size_t)(b * C + g * chPerG)) * (size_t)SP;
  const size_t T = (size_t)chPerG * (size_t)SP;
  float s = 0.f, q = 0.f;
  for (size_t i = threadIdx.x; i < T; i += 256) {
    float v = x[base + i];
    s += v; q += v * v;
  }
  __shared__ float sh0[256], sh1[256];
  sh0[threadIdx.x] = s; sh1[threadIdx.x] = q;
  __syncthreads();
  for (int st = 128; st > 0; st >>= 1) {
    if ((int)threadIdx.x < st) {
      sh0[threadIdx.x] += sh0[threadIdx.x + st];
      sh1[threadIdx.x] += sh1[threadIdx.x + st];
    }
    __syncthreads();
  }
  if (threadIdx.x == 0) {
    float inv = 1.0f / (float)T;
    float mean = sh0[0] * inv;
    float var = sh1[0] * inv - mean * mean;
    stats[2 * bg]     = mean;
    stats[2 * bg + 1] = rsqrtf(var + 1e-5f);
  }
}

__global__ void gn_apply(float* __restrict__ y, const float* __restrict__ gamma,
                         const float* __restrict__ beta, const float* __restrict__ stats,
                         int C, int SP, int chPerG, int G, long total) {
  long i = (long)blockIdx.x * blockDim.x + threadIdx.x;
  if (i >= total) return;
  long t = i / SP;
  int c = (int)(t % C);
  int b = (int)(t / C);
  int bg = b * G + c / chPerG;
  float v = (y[i] - stats[2 * bg]) * stats[2 * bg + 1] * gamma[c] + beta[c];
  y[i] = v > 0.f ? v : 0.f;
}

// Bilinear (align_corners=True) over the two dims preceding a contiguous tail of SS
// elements; optionally adds `addb` (fuses the skip-sum). SS=1 => plain 2D bilinear.
__global__ void interp_add(const float* __restrict__ src, const float* __restrict__ addb,
                           float* __restrict__ out, int C, int Hin, int Win,
                           int Hout, int Wout, int SS, float sy, float sx, long total) {
  long i = (long)blockIdx.x * blockDim.x + threadIdx.x;
  if (i >= total) return;
  long t = i / SS; int ss = (int)(i - t * SS);
  int q = (int)(t % Wout); t /= Wout;
  int pp = (int)(t % Hout); t /= Hout;
  int c = (int)(t % C);    int b = (int)(t / C);
  float fy = pp * sy, fx = q * sx;
  int i0 = (int)fy; if (i0 > Hin - 2) i0 = Hin - 2;
  int j0 = (int)fx; if (j0 > Win - 2) j0 = Win - 2;
  float wy = fy - i0, wx = fx - j0;
  size_t basei = (((size_t)(b * C + c) * Hin + i0) * Win + j0) * SS + ss;
  size_t rs = (size_t)Win * SS;
  float v00 = src[basei], v01 = src[basei + SS], v10 = src[basei + rs], v11 = src[basei + rs + SS];
  float v = (1.f - wy) * ((1.f - wx) * v00 + wx * v01) + wy * ((1.f - wx) * v10 + wx * v11);
  if (addb) v += addb[i];
  out[i] = v;
}

__global__ void mean_tail(const float* __restrict__ x, float* __restrict__ y, int SS, long n) {
  long i = (long)blockIdx.x * blockDim.x + threadIdx.x;
  if (i >= n) return;
  float s = 0.f;
  for (int j = 0; j < SS; ++j) s += x[(size_t)i * SS + j];
  y[i] = s / (float)SS;
}

template<int NT, int MODE>
static void launch_conv_k(int K, int grid, hipStream_t s, const float* x,
                          const _Float16* w, const float* bias, float* y, const ConvDesc& d) {
  if (K == 5) conv_wmma<NT, MODE, 5><<<grid, 256, 0, s>>>(x, w, bias, y, d);
  else        conv_wmma<NT, MODE, 3><<<grid, 256, 0, s>>>(x, w, bias, y, d);
}
static void launch_conv(int NT, int mode, int K, int grid, hipStream_t s, const float* x,
                        const _Float16* w, const float* bias, float* y, const ConvDesc& d) {
  if (mode == 0) {
    if (NT == 1)      launch_conv_k<1,0>(K, grid, s, x, w, bias, y, d);
    else if (NT == 2) launch_conv_k<2,0>(K, grid, s, x, w, bias, y, d);
    else              launch_conv_k<4,0>(K, grid, s, x, w, bias, y, d);
  } else {
    if (NT == 1)      launch_conv_k<1,1>(K, grid, s, x, w, bias, y, d);
    else if (NT == 2) launch_conv_k<2,1>(K, grid, s, x, w, bias, y, d);
    else              launch_conv_k<4,1>(K, grid, s, x, w, bias, y, d);
  }
}

extern "C" void kernel_launch(void* const* d_in, const int* in_sizes, int n_in,
                              void* d_out, int out_size, void* d_ws, size_t ws_size,
                              hipStream_t stream) {
  (void)in_sizes; (void)n_in; (void)out_size; (void)ws_size;

  const float* corr4 = (const float*)d_in[0];
  const float* corr3 = (const float*)d_in[1];
  const float* corr2 = (const float*)d_in[2];

  struct Lay {
    const float *w1,*b1,*w2,*b2,*g,*bb;
    const _Float16 *w1h,*w2h;
    int cin, cout, k, s, Kp;
  };
  Lay enc4[3], enc3[3], enc2[3], enc43[3], enc32[3];

  int p = 3;
  auto nextp = [&]() -> const float* { return (const float*)d_in[p++]; };
  auto grabBlock = [&](Lay* L, int inch, const int* outs, const int* ks, const int* st) {
    for (int i = 0; i < 3; ++i) {
      L[i].w1 = nextp(); L[i].b1 = nextp(); L[i].w2 = nextp(); L[i].b2 = nextp();
      L[i].g = nextp();  L[i].bb = nextp();
      L[i].cin = inch; L[i].cout = outs[i]; L[i].k = ks[i]; L[i].s = st[i];
      inch = outs[i];
    }
  };
  const int oA[3]={16,32,64}, oB[3]={64,64,64};
  const int k4[3]={3,3,3}, k3_[3]={5,3,3}, k2_[3]={5,5,3}, k1[3]={3,3,3};
  const int st4[3]={2,2,2}, st3[3]={4,2,2}, st2[3]={4,4,2}, st1[3]={1,1,1};
  grabBlock(enc4, 3, oA, k4, st4);
  grabBlock(enc3, 6, oA, k3_, st3);
  grabBlock(enc2, 4, oA, k2_, st2);
  grabBlock(enc43, 64, oB, k1, st1);
  grabBlock(enc32, 64, oB, k1, st1);
  const float* d1w1 = nextp(); const float* d1b1 = nextp();
  const float* d1w2 = nextp(); const float* d1b2 = nextp();
  const float* d2w1 = nextp(); const float* d2b1 = nextp();
  const float* d2w2 = nextp(); const float* d2b2 = nextp();

  // ---- bump arena over d_ws ----
  char* basep = (char*)d_ws;
  size_t off = 0;
  auto alloc = [&](size_t nbytes) -> void* {
    off = (off + 255) & ~(size_t)255;
    void* r = basep + off;
    off += nbytes;
    return r;
  };
  // pack weights to padded [NT*16, Kp] f16 (Kp multiple of 32 => B loads are aligned 32B)
  auto packW = [&](const float* src, int OC, int Ktot, int& KpOut) -> const _Float16* {
    int Kp = (Ktot + 31) & ~31;
    int OCp = ((OC + 15) / 16) * 16;
    int total = OCp * Kp;
    _Float16* dst = (_Float16*)alloc((size_t)total * sizeof(_Float16));
    pack_w_f16<<<(total + 255) / 256, 256, 0, stream>>>(src, dst, OC, Ktot, Kp, total);
    KpOut = Kp;
    return dst;
  };

  float* stats = (float*)alloc(64 * sizeof(float));
  Lay* blocks[5] = {enc4, enc3, enc2, enc43, enc32};
  for (int bix = 0; bix < 5; ++bix)
    for (int i = 0; i < 3; ++i) {
      Lay& L = blocks[bix][i];
      int Ktot = L.cin * L.k * L.k;
      L.w1h = packW(L.w1, L.cout, Ktot, L.Kp);
      L.w2h = packW(L.w2, L.cout, Ktot, L.Kp);
    }
  int kpD1, kpD2, kpD3, kpD4;
  const _Float16* d1w1h = packW(d1w1, 64, 128*9, kpD1);
  const _Float16* d1w2h = packW(d1w2, 32,  64*9, kpD2);
  const _Float16* d2w1h = packW(d2w1, 16,  32*9, kpD3);
  const _Float16* d2w2h = packW(d2w2,  2,  16*9, kpD4);

  float* ps2 = (float*)alloc(1280000ull * 4);   // [2,64,50,50,2,2]
  float* ps3 = (float*)alloc(320000ull * 4);    // [2,64,25,25,2,2]
  float* ps4 = (float*)alloc(86528ull * 4);     // [2,64,13,13,2,2]
  const size_t mark = off;                      // scratch region rewinds here per phase

  // one CenterPivot-4D layer: conv1(query) store, conv2(support) accumulate, GN+ReLU in place
  auto run_layer = [&](const float* x, float* y, const Lay& L,
                       int B, int HA, int WA, int HB, int WB, int& HB2, int& WB2) {
    HB2 = (HB + L.s - 1) / L.s;
    WB2 = (WB + L.s - 1) / L.s;
    ConvDesc dd{};
    dd.B = B; dd.C = L.cin; dd.CR = L.cin;
    dd.HA = HA; dd.WA = WA; dd.HB = HB; dd.WB = WB;
    dd.OC = L.cout; dd.S = L.s; dd.P = L.k / 2;
    dd.HB2 = HB2; dd.WB2 = WB2;
    dd.Mtotal = B * HB2 * WB2 * HA * WA;
    dd.Ktot = L.cin * L.k * L.k;
    dd.Kp = L.Kp;
    dd.relu = 0; dd.accumulate = 0;
    const int NT = (L.cout + 15) / 16;
    const int grid = (dd.Mtotal + 16 * WAVES_PER_BLOCK - 1) / (16 * WAVES_PER_BLOCK);
    launch_conv(NT, 0, L.k, grid, stream, x, L.w1h, L.b1, y, dd);   // out1 = conv1 + b1
    dd.accumulate = 1;
    launch_conv(NT, 1, L.k, grid, stream, x, L.w2h, L.b2, y, dd);   // y += conv2 + b2
    const int SP = HA * WA * HB2 * WB2;
    const int chG = L.cout / NGROUPS;
    gn_stats<<<B * NGROUPS, 256, 0, stream>>>(y, stats, L.cout, SP, chG, NGROUPS);
    const long tot = (long)B * L.cout * SP;
    gn_apply<<<(int)((tot + 255) / 256), 256, 0, stream>>>(y, L.g, L.bb, stats,
                                                           L.cout, SP, chG, NGROUPS, tot);
  };
  auto run_conv2d = [&](const float* x, float* y, const _Float16* wh, int Kp, const float* bias,
                        int B, int Cin, int CR, int OC, int H, int W, int relu) {
    ConvDesc dd{};
    dd.B = B; dd.C = Cin; dd.CR = CR;
    dd.HA = H; dd.WA = W; dd.HB = 1; dd.WB = 1;
    dd.OC = OC; dd.S = 1; dd.P = 1; dd.HB2 = 1; dd.WB2 = 1;
    dd.Mtotal = B * H * W; dd.Ktot = Cin * 9; dd.Kp = Kp;
    dd.relu = relu; dd.accumulate = 0;
    const int NT = (OC + 15) / 16;
    const int grid = (dd.Mtotal + 127) / 128;
    launch_conv(NT, 0, 3, grid, stream, x, wh, bias, y, dd);
  };

  int hb2a, wb2a, hb2b, wb2b, hb2c, wb2c;

  // ---- enc2 (dominant: 200 MB input, 54 MB layer-1 activation) ----
  {
    off = mark;
    float* y0 = (float*)alloc(13520000ull * 4);  // [2,16,50,50,13,13]
    float* y1 = (float*)alloc(2560000ull * 4);   // [2,32,50,50,4,4]
    run_layer(corr2, y0, enc2[0], 2, 50, 50, 50, 50, hb2a, wb2a);
    run_layer(y0,   y1, enc2[1], 2, 50, 50, hb2a, wb2a, hb2b, wb2b);
    run_layer(y1,  ps2, enc2[2], 2, 50, 50, hb2b, wb2b, hb2c, wb2c);
  }
  // ---- enc3 ----
  {
    off = mark;
    float* y0 = (float*)alloc(980000ull * 4);
    float* y1 = (float*)alloc(640000ull * 4);
    run_layer(corr3, y0, enc3[0], 2, 25, 25, 25, 25, hb2a, wb2a);
    run_layer(y0,   y1, enc3[1], 2, 25, 25, hb2a, wb2a, hb2b, wb2b);
    run_layer(y1,  ps3, enc3[2], 2, 25, 25, hb2b, wb2b, hb2c, wb2c);
  }
  // ---- enc4 ----
  {
    off = mark;
    float* y0 = (float*)alloc(264992ull * 4);
    float* y1 = (float*)alloc(173056ull * 4);
    run_layer(corr4, y0, enc4[0], 2, 13, 13, 13, 13, hb2a, wb2a);
    run_layer(y0,   y1, enc4[1], 2, 13, 13, hb2a, wb2a, hb2b, wb2b);
    run_layer(y1,  ps4, enc4[2], 2, 13, 13, hb2b, wb2b, hb2c, wb2c);
  }

  // ---- mixers (support dims are 2x2 everywhere below; SS = 4) ----
  off = mark;
  float* t0 = (float*)alloc(320000ull * 4);
  float* t1 = (float*)alloc(320000ull * 4);
  float* t2 = (float*)alloc(320000ull * 4);
  {
    long n = 2L * 64 * 25 * 25 * 4;
    interp_add<<<(int)((n + 255) / 256), 256, 0, stream>>>(
        ps4, ps3, t0, 64, 13, 13, 25, 25, 4, 12.f / 24.f, 12.f / 24.f, n);
    run_layer(t0, t1, enc43[0], 2, 25, 25, 2, 2, hb2a, wb2a);
    run_layer(t1, t2, enc43[1], 2, 25, 25, 2, 2, hb2a, wb2a);
    run_layer(t2, t0, enc43[2], 2, 25, 25, 2, 2, hb2a, wb2a);   // m43 -> t0
  }
  float* u0 = (float*)alloc(1280000ull * 4);
  float* u1 = (float*)alloc(1280000ull * 4);
  float* u2 = (float*)alloc(1280000ull * 4);
  {
    long n = 2L * 64 * 50 * 50 * 4;
    interp_add<<<(int)((n + 255) / 256), 256, 0, stream>>>(
        t0, ps2, u0, 64, 25, 25, 50, 50, 4, 24.f / 49.f, 24.f / 49.f, n);
    run_layer(u0, u1, enc32[0], 2, 50, 50, 2, 2, hb2a, wb2a);
    run_layer(u1, u2, enc32[1], 2, 50, 50, 2, 2, hb2a, wb2a);
    run_layer(u2, u0, enc32[2], 2, 50, 50, 2, 2, hb2a, wb2a);   // m432 -> u0
  }

  // ---- mean over support + decoder (concat handled by channel wrap CR=64, C=128) ----
  float* e = (float*)alloc(320000ull * 4);                      // [2,64,50,50]
  {
    long n = 2L * 64 * 2500;
    mean_tail<<<(int)((n + 255) / 256), 256, 0, stream>>>(u0, e, 4, n);
  }
  float* h1 = (float*)alloc(320000ull * 4);
  run_conv2d(e, h1, d1w1h, kpD1, d1b1, 2, 128, 64, 64, 50, 50, 1);
  float* h2 = (float*)alloc(160000ull * 4);
  run_conv2d(h1, h2, d1w2h, kpD2, d1b2, 2, 64, 64, 32, 50, 50, 1);
  float* up = (float*)alloc(640000ull * 4);                     // [2,32,100,100]
  {
    long n = 2L * 32 * 100 * 100;
    interp_add<<<(int)((n + 255) / 256), 256, 0, stream>>>(
        h2, nullptr, up, 32, 50, 50, 100, 100, 1, 49.f / 99.f, 49.f / 99.f, n);
  }
  float* h3 = (float*)alloc(320000ull * 4);
  run_conv2d(up, h3, d2w1h, kpD3, d2b1, 2, 32, 32, 16, 100, 100, 1);
  run_conv2d(h3, (float*)d_out, d2w2h, kpD4, d2b2, 2, 16, 16, 2, 100, 100, 0);  // [2,2,100,100]
}
